// Ernie4_5_VLMoeMoE_9294309228908
// MI455X (gfx1250) — matrix operand order repaired and checked by reading, not compiled, not run on hardware
//
#include <hip/hip_runtime.h>
#include <cstdint>
#include <cstddef>

// ---------------------------------------------------------------------------
// Ernie4.5-VL MoE layer for gfx1250 (MI455X), wave32 + WMMA bf16.
// bf16 convert -> router top-6 (active branch only) -> gather lists
// -> grouped GEMM1 (w13, fused SwiGLU) -> grouped GEMM2 (w2, scaled scatter-add)
// A tiles staged with GLOBAL_LOAD_ASYNC_TO_LDS_B128 (ASYNCcnt), weights
// converted fp32->bf16 while streaming, double-buffered LDS pipeline.
// ---------------------------------------------------------------------------

#define T_TOK    4096
#define H_DIM    1024
#define NE       64          // experts per branch
#define NEG      128         // global experts (text 0..63, vision 64..127)
#define TOPK     6
#define I_TEXT   512
#define I_VIS    256
#define I_SHARED 1024        // I_TEXT * n_shared(2)
#define MAXROWS  (T_TOK * TOPK)

typedef __bf16 bf16;
typedef __attribute__((ext_vector_type(16))) __bf16 v16bf;
typedef __attribute__((ext_vector_type(8)))  __bf16 v8bf;
typedef __attribute__((ext_vector_type(4)))  __bf16 v4bf;
typedef __attribute__((ext_vector_type(8)))  float  v8f;

__device__ __forceinline__ v8f zero_v8f() {
    v8f z = {0.f, 0.f, 0.f, 0.f, 0.f, 0.f, 0.f, 0.f};
    return z;
}

// fast SiLU: g * 1/(1+exp(-g)) -> v_exp_f32 + v_rcp_f32
__device__ __forceinline__ float fast_silu(float g) {
    return g * __builtin_amdgcn_rcpf(1.0f + __expf(-g));
}

// CDNA5 async copy: 16 bytes global -> LDS, tracked with ASYNCcnt.
__device__ __forceinline__ void async_b128_to_lds(void* lds_dst, const void* gsrc) {
    uint32_t lds_addr = (uint32_t)(uintptr_t)lds_dst;
    asm volatile("global_load_async_to_lds_b128 %0, %1, off"
                 :: "v"(lds_addr), "v"(gsrc)
                 : "memory");
}
__device__ __forceinline__ void wait_async_zero() {
    asm volatile("s_wait_asynccnt 0x0" ::: "memory");
}

// ---------------------------------------------------------------------------
// 0) fp32 -> bf16 activation convert
// ---------------------------------------------------------------------------
__global__ void __launch_bounds__(256) cvt_bf16_kernel(const float* __restrict__ x,
                                                       bf16* __restrict__ xbf, int n) {
    int i = (blockIdx.x * 256 + threadIdx.x) * 4;
    if (i < n) {
        float4 v = *(const float4*)(x + i);
        v4bf b = {(bf16)v.x, (bf16)v.y, (bf16)v.z, (bf16)v.w};
        *(v4bf*)(xbf + i) = b;
    }
}

// ---------------------------------------------------------------------------
// 1) Router: one block (64 threads) per token; only the mask-active branch.
// ---------------------------------------------------------------------------
__global__ void __launch_bounds__(64) router_topk_kernel(
    const float* __restrict__ x, const unsigned char* __restrict__ vmask,
    const float* __restrict__ ebias, const float* __restrict__ tgate,
    const float* __restrict__ vgate,
    int* __restrict__ topk_idx, float* __restrict__ topk_w) {
    __shared__ float xs[H_DIM];
    __shared__ float probs[NE];
    __shared__ float sc[NE];

    const int t = blockIdx.x;
    const int tid = threadIdx.x;
    const int vis = vmask[t] ? 1 : 0;
    const float* gw = vis ? vgate : tgate;
    const float* bias = ebias + vis * NE;

    for (int i = tid; i < H_DIM; i += 64) xs[i] = x[(size_t)t * H_DIM + i];
    __syncthreads();

    float acc = 0.f;
    const float* wr = gw + (size_t)tid * H_DIM;
    for (int k = 0; k < H_DIM; k += 4) {
        float4 wv = *(const float4*)(wr + k);
        acc += xs[k] * wv.x + xs[k + 1] * wv.y + xs[k + 2] * wv.z + xs[k + 3] * wv.w;
    }
    probs[tid] = acc;
    __syncthreads();

    if (tid == 0) {
        float mx = probs[0];
        for (int i = 1; i < NE; ++i) mx = fmaxf(mx, probs[i]);
        float s = 0.f;
        for (int i = 0; i < NE; ++i) { float p = __expf(probs[i] - mx); probs[i] = p; s += p; }
        float inv = __builtin_amdgcn_rcpf(s);
        for (int i = 0; i < NE; ++i) { probs[i] *= inv; sc[i] = probs[i] + bias[i]; }
        int sel[TOPK]; float selw[TOPK]; float wsum = 0.f;
        for (int j = 0; j < TOPK; ++j) {
            int im = 0; float bm = sc[0];
            for (int i = 1; i < NE; ++i) if (sc[i] > bm) { bm = sc[i]; im = i; }
            sel[j] = im; selw[j] = probs[im]; wsum += probs[im]; sc[im] = -1e30f;
        }
        float invw = __builtin_amdgcn_rcpf(wsum);
        for (int j = 0; j < TOPK; ++j) {
            topk_idx[t * TOPK + j] = vis * NE + sel[j];
            topk_w[t * TOPK + j] = selw[j] * invw;
        }
    }
}

// ---------------------------------------------------------------------------
// 2) count / scan / scatter  ->  per-expert contiguous token lists
// ---------------------------------------------------------------------------
__global__ void __launch_bounds__(256) moe_count_kernel(const int* __restrict__ topk_idx,
                                                        int* __restrict__ counts, int n) {
    int i = blockIdx.x * 256 + threadIdx.x;
    if (i < n) atomicAdd(&counts[topk_idx[i]], 1);
}

__global__ void moe_scan_kernel(const int* __restrict__ counts,
                                int* __restrict__ offsets, int* __restrict__ cursor) {
    if (threadIdx.x == 0 && blockIdx.x == 0) {
        int acc = 0;
        for (int i = 0; i < NEG; ++i) { offsets[i] = acc; cursor[i] = acc; acc += counts[i]; }
    }
}

__global__ void __launch_bounds__(256) moe_scatter_kernel(
    const int* __restrict__ topk_idx, const float* __restrict__ topk_w,
    int* __restrict__ cursor, int* __restrict__ rowids, float* __restrict__ rweight, int n) {
    int i = blockIdx.x * 256 + threadIdx.x;
    if (i < n) {
        int g = topk_idx[i];
        int r = atomicAdd(&cursor[g], 1);
        rowids[r] = i / TOPK;
        rweight[r] = topk_w[i];
    }
}

// ---------------------------------------------------------------------------
// 3) Grouped GEMM1: h = silu(x @ Wg^T) * (x @ Wu^T)   (bf16 WMMA, f32 acc)
//    64x64 workgroup tile, 8 waves = 4M x 2N, 2 N-subtiles per wave.
//    Double-buffered LDS; A staged async-to-LDS; constexpr strides.
// ---------------------------------------------------------------------------
template <int IN, int LDH>
__global__ void __launch_bounds__(256) moe_gemm1_silu_kernel(
    const bf16* __restrict__ xbf,
    const int* __restrict__ rowids, const int* __restrict__ offsets,
    const int* __restrict__ counts,
    const float* __restrict__ w13,  // [E][2*IN][H] (or [2*IN][H] when rowids==null)
    int ebase, int fixedCount,
    bf16* __restrict__ hout) {
    __shared__ __attribute__((aligned(32))) bf16 As[2][64 * 32];
    __shared__ __attribute__((aligned(32))) bf16 Bg[2][64 * 32];
    __shared__ __attribute__((aligned(32))) bf16 Bu[2][64 * 32];

    const int tid = threadIdx.x;
    const int lane = tid & 31;
    const int wave = tid >> 5;
    const int wm = wave >> 1;   // 0..3 (M)
    const int wn = wave & 1;    // 0..1 (N), 2 subtiles each
    const int e = blockIdx.y;
    const int n0 = blockIdx.x * 64;

    int base, count;
    const float* w;
    if (rowids) {
        base = offsets[ebase + e];
        count = counts[ebase + e];
        w = w13 + (size_t)e * (size_t)(2 * IN) * H_DIM;
    } else {
        base = 0; count = fixedCount; w = w13;
    }
    if (count <= 0) return;

    const int ar = tid >> 2;        // A stage row 0..63
    const int ak = (tid & 3) * 8;   // A stage k {0,8,16,24} (16B)
    constexpr int KITER = H_DIM / 32;   // 32 (even -> buffer parity safe)

    for (int m0 = 0; m0 < count; m0 += 64) {
        v8f accg[2] = {zero_v8f(), zero_v8f()};
        v8f accu[2] = {zero_v8f(), zero_v8f()};

        int am = m0 + ar;
        int amc = am < count ? am : count - 1;          // clamp: masked later
        int srow = rowids ? rowids[base + amc] : amc;
        const bf16* asrc = xbf + (size_t)srow * H_DIM;

        auto stage = [&](int b, int k0) {
            async_b128_to_lds(&As[b][ar * 32 + ak], asrc + k0 + ak);
#pragma unroll
            for (int it = 0; it < 2; ++it) {
                int idx = it * 256 + tid;
                int rn = idx >> 3;
                int kk = (idx & 7) * 4;
                const float* gp = w + (size_t)(n0 + rn) * H_DIM + k0 + kk;
                const float* up = w + (size_t)(IN + n0 + rn) * H_DIM + k0 + kk;
                __builtin_prefetch(gp + 32, 0, 1);
                __builtin_prefetch(up + 32, 0, 1);
                float4 gv = *(const float4*)gp;
                float4 uv = *(const float4*)up;
                v4bf gb = {(bf16)gv.x, (bf16)gv.y, (bf16)gv.z, (bf16)gv.w};
                v4bf ub = {(bf16)uv.x, (bf16)uv.y, (bf16)uv.z, (bf16)uv.w};
                *(v4bf*)(&Bg[b][rn * 32 + kk]) = gb;
                *(v4bf*)(&Bu[b][rn * 32 + kk]) = ub;
            }
        };

        stage(0, 0);  // prologue (touches buf0; trailing compute of prev m0 used buf1)
        for (int ki = 0; ki < KITER; ++ki) {
            wait_async_zero();
            __syncthreads();
            if (ki + 1 < KITER) stage((ki + 1) & 1, (ki + 1) * 32);
            const int b = ki & 1;
            const int arow = wm * 16 + (lane & 15);
            const int ksel = (lane >> 4) * 16;
            v8bf alo = *(const v8bf*)(&As[b][arow * 32 + ksel]);
            v8bf ahi = *(const v8bf*)(&As[b][arow * 32 + ksel + 8]);
            v16bf afrag;
#pragma unroll
            for (int i = 0; i < 8; ++i) { afrag[i] = alo[i]; afrag[i + 8] = ahi[i]; }
            const int bk = (lane >> 4) * 16;
#pragma unroll
            for (int s = 0; s < 2; ++s) {
                const int brow = wn * 32 + s * 16 + (lane & 15);
                v16bf bg = *(const v16bf*)(&Bg[b][brow * 32 + bk]);
                v16bf bu = *(const v16bf*)(&Bu[b][brow * 32 + bk]);
                accg[s] = __builtin_amdgcn_wmma_f32_16x16x32_bf16(
                    false, afrag, false, bg, (short)0, accg[s], false, false);
                accu[s] = __builtin_amdgcn_wmma_f32_16x16x32_bf16(
                    false, afrag, false, bu, (short)0, accu[s], false, false);
            }
        }
        // epilogue: SwiGLU fused; constexpr LDH folds r*LDH into store offsets
        const int mb = m0 + wm * 16 + (lane >> 4) * 8;
        bf16* hp0 = hout + (size_t)(base + mb) * LDH;
#pragma unroll
        for (int s = 0; s < 2; ++s) {
            bf16* hp = hp0 + n0 + wn * 32 + s * 16 + (lane & 15);
#pragma unroll
            for (int r = 0; r < 8; ++r) {
                if (mb + r < count) {
                    float h = fast_silu(accg[s][r]) * accu[s][r];
                    hp[(size_t)r * LDH] = (bf16)h;
                }
            }
        }
    }
}

// ---------------------------------------------------------------------------
// 4) Grouped GEMM2: out += weight * (h @ W2^T)   (atomic scatter-add)
// ---------------------------------------------------------------------------
template <int KD, int LDH, bool ATOMIC>
__global__ void __launch_bounds__(256) moe_gemm2_scatter_kernel(
    const bf16* __restrict__ hin,
    const int* __restrict__ rowids, const int* __restrict__ offsets,
    const int* __restrict__ counts,
    const float* __restrict__ w2,  // [E][H][KD] (or [H][KD] when rowids==null)
    int ebase, int fixedCount,
    const float* __restrict__ rweight,
    float* __restrict__ out) {
    __shared__ __attribute__((aligned(32))) bf16 As[2][64 * 32];
    __shared__ __attribute__((aligned(32))) bf16 Bs[2][64 * 32];

    const int tid = threadIdx.x;
    const int lane = tid & 31;
    const int wave = tid >> 5;
    const int wm = wave >> 1;
    const int wn = wave & 1;
    const int e = blockIdx.y;
    const int n0 = blockIdx.x * 64;

    int base, count;
    const float* w;
    if (rowids) {
        base = offsets[ebase + e];
        count = counts[ebase + e];
        w = w2 + (size_t)e * (size_t)H_DIM * KD;
    } else {
        base = 0; count = fixedCount; w = w2;
    }
    if (count <= 0) return;

    const int ar = tid >> 2;
    const int ak = (tid & 3) * 8;
    constexpr int KITER = KD / 32;   // 8/16/32, all even

    for (int m0 = 0; m0 < count; m0 += 64) {
        v8f acc[2] = {zero_v8f(), zero_v8f()};

        int am = m0 + ar;
        int amc = am < count ? am : count - 1;
        const bf16* asrc = hin + (size_t)(base + amc) * LDH;

        auto stage = [&](int b, int k0) {
            async_b128_to_lds(&As[b][ar * 32 + ak], asrc + k0 + ak);
#pragma unroll
            for (int it = 0; it < 2; ++it) {
                int idx = it * 256 + tid;
                int rn = idx >> 3;
                int kk = (idx & 7) * 4;
                const float* bp = w + (size_t)(n0 + rn) * KD + k0 + kk;
                __builtin_prefetch(bp + 32, 0, 1);
                float4 bv = *(const float4*)bp;
                v4bf bb = {(bf16)bv.x, (bf16)bv.y, (bf16)bv.z, (bf16)bv.w};
                *(v4bf*)(&Bs[b][rn * 32 + kk]) = bb;
            }
        };

        stage(0, 0);
        for (int ki = 0; ki < KITER; ++ki) {
            wait_async_zero();
            __syncthreads();
            if (ki + 1 < KITER) stage((ki + 1) & 1, (ki + 1) * 32);
            const int b = ki & 1;
            const int arow = wm * 16 + (lane & 15);
            const int ksel = (lane >> 4) * 16;
            v8bf alo = *(const v8bf*)(&As[b][arow * 32 + ksel]);
            v8bf ahi = *(const v8bf*)(&As[b][arow * 32 + ksel + 8]);
            v16bf afrag;
#pragma unroll
            for (int i = 0; i < 8; ++i) { afrag[i] = alo[i]; afrag[i + 8] = ahi[i]; }
            const int bk = (lane >> 4) * 16;
#pragma unroll
            for (int s = 0; s < 2; ++s) {
                const int brow = wn * 32 + s * 16 + (lane & 15);
                v16bf bfrag = *(const v16bf*)(&Bs[b][brow * 32 + bk]);
                acc[s] = __builtin_amdgcn_wmma_f32_16x16x32_bf16(
                    false, afrag, false, bfrag, (short)0, acc[s], false, false);
            }
        }
        const int mb = m0 + wm * 16 + (lane >> 4) * 8;
#pragma unroll
        for (int s = 0; s < 2; ++s) {
            const int n = n0 + wn * 32 + s * 16 + (lane & 15);
            if (ATOMIC) {
#pragma unroll
                for (int r = 0; r < 8; ++r) {
                    int m = mb + r;
                    if (m < count) {
                        int tok = rowids[base + m];
                        atomicAdd(out + (size_t)tok * H_DIM + n,
                                  acc[s][r] * rweight[base + m]);
                    }
                }
            } else {
                float* op = out + (size_t)mb * H_DIM + n;
#pragma unroll
                for (int r = 0; r < 8; ++r) {
                    if (mb + r < count) op[(size_t)r * H_DIM] = acc[s][r];
                }
            }
        }
    }
}

// ---------------------------------------------------------------------------
// Launch
// ---------------------------------------------------------------------------
extern "C" void kernel_launch(void* const* d_in, const int* in_sizes, int n_in,
                              void* d_out, int out_size, void* d_ws, size_t ws_size,
                              hipStream_t stream) {
    const float* x = (const float*)d_in[0];
    const unsigned char* vmask = (const unsigned char*)d_in[1];
    const float* ebias = (const float*)d_in[2];
    const float* tgate = (const float*)d_in[3];
    const float* vgate = (const float*)d_in[4];
    const float* text_w13 = (const float*)d_in[5];
    const float* text_w2 = (const float*)d_in[6];
    const float* vis_w13 = (const float*)d_in[7];
    const float* vis_w2 = (const float*)d_in[8];
    const float* sh_w13 = (const float*)d_in[9];
    const float* sh_w2 = (const float*)d_in[10];
    float* out = (float*)d_out;

    uintptr_t wbase = (uintptr_t)d_ws;
    size_t off = 0;
    auto carve = [&](size_t bytes) -> void* {
        off = (off + 255) & ~(size_t)255;
        void* p = (void*)(wbase + off);
        off += bytes;
        return p;
    };
    bf16* xbf      = (bf16*)carve((size_t)T_TOK * H_DIM * sizeof(bf16));
    bf16* hshared  = (bf16*)carve((size_t)T_TOK * I_SHARED * sizeof(bf16));
    bf16* hbuf     = (bf16*)carve((size_t)MAXROWS * I_TEXT * sizeof(bf16));
    int* topk_idx  = (int*)carve((size_t)T_TOK * TOPK * sizeof(int));
    float* topk_w  = (float*)carve((size_t)T_TOK * TOPK * sizeof(float));
    int* counts    = (int*)carve(NEG * sizeof(int));
    int* offsets   = (int*)carve(NEG * sizeof(int));
    int* cursor    = (int*)carve(NEG * sizeof(int));
    int* rowids    = (int*)carve((size_t)MAXROWS * sizeof(int));
    float* rweight = (float*)carve((size_t)MAXROWS * sizeof(float));
    (void)ws_size; (void)in_sizes; (void)n_in; (void)out_size;

    // 0) bf16 activations
    cvt_bf16_kernel<<<(T_TOK * H_DIM) / 1024, 256, 0, stream>>>(x, xbf, T_TOK * H_DIM);
    // 1) router (active branch only)
    router_topk_kernel<<<T_TOK, 64, 0, stream>>>(x, vmask, ebias, tgate, vgate,
                                                 topk_idx, topk_w);
    // 2) gather lists
    hipMemsetAsync(counts, 0, NEG * sizeof(int), stream);
    moe_count_kernel<<<(T_TOK * TOPK + 255) / 256, 256, 0, stream>>>(topk_idx, counts,
                                                                     T_TOK * TOPK);
    moe_scan_kernel<<<1, 32, 0, stream>>>(counts, offsets, cursor);
    moe_scatter_kernel<<<(T_TOK * TOPK + 255) / 256, 256, 0, stream>>>(
        topk_idx, topk_w, cursor, rowids, rweight, T_TOK * TOPK);

    // 3) shared experts: dense SwiGLU -> plain store into out (runs first)
    moe_gemm1_silu_kernel<I_SHARED, I_SHARED><<<dim3(I_SHARED / 64, 1), 256, 0, stream>>>(
        xbf, nullptr, nullptr, nullptr, sh_w13, 0, T_TOK, hshared);
    moe_gemm2_scatter_kernel<I_SHARED, I_SHARED, false>
        <<<dim3(H_DIM / 64, 1), 256, 0, stream>>>(
        hshared, nullptr, nullptr, nullptr, sh_w2, 0, T_TOK, nullptr, out);

    // 4) text branch (global experts 0..63)
    moe_gemm1_silu_kernel<I_TEXT, I_TEXT><<<dim3(I_TEXT / 64, NE), 256, 0, stream>>>(
        xbf, rowids, offsets, counts, text_w13, 0, 0, hbuf);
    moe_gemm2_scatter_kernel<I_TEXT, I_TEXT, true>
        <<<dim3(H_DIM / 64, NE), 256, 0, stream>>>(
        hbuf, rowids, offsets, counts, text_w2, 0, 0, rweight, out);

    // 5) vision branch (global experts 64..127), shares hbuf with stride I_TEXT
    moe_gemm1_silu_kernel<I_VIS, I_TEXT><<<dim3(I_VIS / 64, NE), 256, 0, stream>>>(
        xbf, rowids, offsets, counts, vis_w13, NE, 0, hbuf);
    moe_gemm2_scatter_kernel<I_VIS, I_TEXT, true>
        <<<dim3(H_DIM / 64, NE), 256, 0, stream>>>(
        hbuf, rowids, offsets, counts, vis_w2, NE, 0, rweight, out);
}
